// CLUB_42176578847177
// MI455X (gfx1250) — matrix operand order repaired and checked by reading, not compiled, not run on hardware
//
#include <hip/hip_runtime.h>
#include <hip/hip_bf16.h>
#include <math.h>

typedef __attribute__((ext_vector_type(16))) _Float16 v16h;
typedef __attribute__((ext_vector_type(8)))  _Float16 v8h;
typedef __attribute__((ext_vector_type(8)))  float    v8f;

#define NB  8
#define NL  512
#define NXD 192
#define NYD 64
#define NH  128

// ---- workspace layout (bytes) ----
#define OFF_W1MU 0
#define OFF_W1LV (OFF_W1MU + NH*NXD*2)          // 49152
#define OFF_W2MU (OFF_W1LV + NH*NXD*2)          // 98304
#define OFF_W2LV (OFF_W2MU + NYD*NH*2)          // 114688
#define OFF_XT   (OFF_W2LV + NYD*NH*2)          // 131072
#define OFF_HMU  (OFF_XT   + NB*NL*NXD*2)       // 1703936
#define OFF_HLV  (OFF_HMU  + NB*NL*NH*2)        // 2752512
#define OFF_EY   (OFF_HLV  + NB*NL*NH*2)        // 3801088
#define OFF_EY2  (OFF_EY   + NB*NYD*4)          // 3803136
// total ~3.8 MB

// A-fragment (16x32 f16, M x K) from row-major W[ldk cols], tile origin (m0,k0).
// Per ISA layout: lane<16 -> K base 0, lane>=16 -> K base 8; two contiguous
// 8-half runs at K = base..base+7 and K = 16+base..16+base+7.
__device__ inline v16h load_a_frag(const _Float16* __restrict__ W, int ldk,
                                   int m0, int k0, int lane) {
  int mrow  = lane & 15;
  int abase = (lane & 16) ? 8 : 0;
  const _Float16* p = W + (m0 + mrow) * ldk + k0 + abase;
  union { v16h v; v8h h[2]; } u;
  u.h[0] = *reinterpret_cast<const v8h*>(p);
  u.h[1] = *reinterpret_cast<const v8h*>(p + 16);
  return u.v;
}

// B-fragment (32x16 f16, K x N) from n-major Xt (element (K,N) at Xt[N*ldk+K]),
// tile origin (k0, n0). lane<16 -> K base 0, lane>=16 -> K base 8; two 8-half
// runs at K = base..base+7 and K = 16+base..16+base+7 (matches A).
__device__ inline v16h load_b_frag(const _Float16* __restrict__ Xt, int ldk,
                                   int n0, int k0, int lane) {
  int ncol  = lane & 15;
  int bbase = (lane & 16) ? 8 : 0;
  const _Float16* p = Xt + (n0 + ncol) * ldk + k0 + bbase;
  union { v16h v; v8h h[2]; } u;
  u.h[0] = *reinterpret_cast<const v8h*>(p);
  u.h[1] = *reinterpret_cast<const v8h*>(p + 16);
  return u.v;
}

// ---- prep: weight fp32->fp16 conversion + zero output scalar ----
__global__ void k_prep_weights(const float* __restrict__ W1mu, const float* __restrict__ W1lv,
                               const float* __restrict__ W2mu, const float* __restrict__ W2lv,
                               _Float16* __restrict__ oW1mu, _Float16* __restrict__ oW1lv,
                               _Float16* __restrict__ oW2mu, _Float16* __restrict__ oW2lv,
                               float* __restrict__ out) {
  int t = blockIdx.x * blockDim.x + threadIdx.x;
  if (t == 0) out[0] = 0.0f;
  const int n1 = NH * NXD;   // 24576
  const int n2 = NYD * NH;   // 8192
  if (t < n1)                 oW1mu[t]            = (_Float16)W1mu[t];
  else if (t < 2 * n1)        oW1lv[t - n1]       = (_Float16)W1lv[t - n1];
  else if (t < 2 * n1 + n2)   oW2mu[t - 2 * n1]   = (_Float16)W2mu[t - 2 * n1];
  else if (t < 2*n1 + 2*n2)   oW2lv[t - 2*n1 - n2] = (_Float16)W2lv[t - 2*n1 - n2];
}

// ---- prep: x [B][XD][L] fp32 -> Xt [B][L][XD] fp16 ----
__global__ void k_transpose_x(const float* __restrict__ x, _Float16* __restrict__ Xt) {
  int t = blockIdx.x * blockDim.x + threadIdx.x;   // output index
  int c = t % NXD;
  int l = (t / NXD) % NL;
  int b = t / (NXD * NL);
  Xt[t] = (_Float16)x[(b * NXD + c) * NL + l];
}

// ---- prep: per-(b,d) mean(y) and mean(y^2) over L ----
__global__ void k_ystats(const float* __restrict__ y,
                         float* __restrict__ Ey, float* __restrict__ Ey2) {
  int t = blockIdx.x * blockDim.x + threadIdx.x;   // t = b*NYD + d
  if (t >= NB * NYD) return;
  const float* p = y + t * NL;
  float s = 0.f, s2 = 0.f;
  for (int l = 0; l < NL; ++l) { float v = p[l]; s += v; s2 += v * v; }
  Ey[t]  = s  * (1.0f / NL);
  Ey2[t] = s2 * (1.0f / NL);
}

// ---- GEMM1: h = relu(W1 @ x + b1), both mu and lv paths. One wave per 16x16 tile.
__global__ __launch_bounds__(128) void k_gemm1(
    const _Float16* __restrict__ W1mu, const _Float16* __restrict__ W1lv,
    const float* __restrict__ b1mu,    const float* __restrict__ b1lv,
    const _Float16* __restrict__ Xt,
    _Float16* __restrict__ Hmu, _Float16* __restrict__ Hlv) {
  int w    = blockIdx.x * 4 + (threadIdx.x >> 5);  // 4096 waves
  int lane = threadIdx.x & 31;
  int ht   = w & 7;          // H tile (8)
  int lt   = (w >> 3) & 31;  // L tile (32)
  int b    = (w >> 8) & 7;   // batch (8)
  int path = w >> 11;        // 0 = mu, 1 = lv
  const _Float16* W1 = path ? W1lv : W1mu;
  const float*    b1 = path ? b1lv : b1mu;
  _Float16*       Hh = path ? Hlv  : Hmu;
  int m0 = ht * 16;
  int n0 = b * NL + lt * 16;     // global row in Xt / H
  v8f c = {};
  #pragma unroll
  for (int kk = 0; kk < NXD / 32; ++kk) {
    v16h a  = load_a_frag(W1, NXD, m0, kk * 32, lane);
    v16h bf = load_b_frag(Xt, NXD, n0, kk * 32, lane);
    c = __builtin_amdgcn_wmma_f32_16x16x32_f16(false, a, false, bf,
                                               (short)0, c, false, false);
  }
  // C layout: reg r -> M = m0 + r + (lane>=16 ? 8 : 0), N = n0 + (lane&15).
  // h index is contiguous across r -> single 16-byte store per lane.
  int moff = (lane & 16) ? 8 : 0;
  int ncol = lane & 15;
  v8h hv;
  #pragma unroll
  for (int r = 0; r < 8; ++r) {
    float v = c[r] + b1[m0 + moff + r];
    hv[r] = (_Float16)fmaxf(v, 0.0f);
  }
  *reinterpret_cast<v8h*>(Hh + (n0 + ncol) * NH + m0 + moff) = hv;
}

// ---- GEMM2 + fused epilogue: mu/lv second layer, tanh/exp, closed-form loss,
//      wave reduction, one atomicAdd per wave. One wave per 16x16 (o,l) tile.
__global__ __launch_bounds__(128) void k_gemm2(
    const _Float16* __restrict__ W2mu, const _Float16* __restrict__ W2lv,
    const float* __restrict__ b2mu,    const float* __restrict__ b2lv,
    const _Float16* __restrict__ Hmu,  const _Float16* __restrict__ Hlv,
    const float* __restrict__ y, const float* __restrict__ Ey,
    const float* __restrict__ Ey2, float* __restrict__ out) {
  int w    = blockIdx.x * 4 + (threadIdx.x >> 5);  // 1024 waves
  int lane = threadIdx.x & 31;
  int ot = w & 3;            // YD tile (4)
  int lt = (w >> 2) & 31;    // L tile (32)
  int b  = w >> 7;           // batch (8)
  int m0 = ot * 16;
  int n0 = b * NL + lt * 16;
  v8f cmu = {}, clv = {};
  #pragma unroll
  for (int kk = 0; kk < NH / 32; ++kk) {
    v16h amu = load_a_frag(W2mu, NH, m0, kk * 32, lane);
    v16h bmu = load_b_frag(Hmu, NH, n0, kk * 32, lane);
    cmu = __builtin_amdgcn_wmma_f32_16x16x32_f16(false, amu, false, bmu,
                                                 (short)0, cmu, false, false);
    v16h alv = load_a_frag(W2lv, NH, m0, kk * 32, lane);
    v16h blv = load_b_frag(Hlv, NH, n0, kk * 32, lane);
    clv = __builtin_amdgcn_wmma_f32_16x16x32_f16(false, alv, false, blv,
                                                 (short)0, clv, false, false);
  }
  int moff = (lane & 16) ? 8 : 0;
  int l    = lt * 16 + (lane & 15);
  float part = 0.f;
  #pragma unroll
  for (int r = 0; r < 8; ++r) {
    int o = m0 + moff + r;
    float mu = cmu[r] + b2mu[o];
    float lv = tanhf(clv[r] + b2lv[o]);
    float iv = expf(-lv);
    float yv  = y[(b * NYD + o) * NL + l];
    float ey  = Ey [b * NYD + o];
    float ey2 = Ey2[b * NYD + o];
    // positive - negative = -0.5*iv*( y^2 - E[y^2] - 2*mu*(y - E[y]) )
    part += -0.5f * iv * (yv * yv - ey2 - 2.0f * mu * (yv - ey));
  }
  #pragma unroll
  for (int off = 16; off > 0; off >>= 1)
    part += __shfl_xor(part, off, 32);
  if (lane == 0) atomicAdd(out, part * (1.0f / (NB * NL)));
}

extern "C" void kernel_launch(void* const* d_in, const int* in_sizes, int n_in,
                              void* d_out, int out_size, void* d_ws, size_t ws_size,
                              hipStream_t stream) {
  (void)in_sizes; (void)n_in; (void)out_size; (void)ws_size;
  const float* x    = (const float*)d_in[0];
  const float* y    = (const float*)d_in[1];
  const float* muW1 = (const float*)d_in[2];
  const float* mub1 = (const float*)d_in[3];
  const float* muW2 = (const float*)d_in[4];
  const float* mub2 = (const float*)d_in[5];
  const float* lvW1 = (const float*)d_in[6];
  const float* lvb1 = (const float*)d_in[7];
  const float* lvW2 = (const float*)d_in[8];
  const float* lvb2 = (const float*)d_in[9];
  char* ws = (char*)d_ws;
  _Float16* wW1mu = (_Float16*)(ws + OFF_W1MU);
  _Float16* wW1lv = (_Float16*)(ws + OFF_W1LV);
  _Float16* wW2mu = (_Float16*)(ws + OFF_W2MU);
  _Float16* wW2lv = (_Float16*)(ws + OFF_W2LV);
  _Float16* Xt    = (_Float16*)(ws + OFF_XT);
  _Float16* Hmu   = (_Float16*)(ws + OFF_HMU);
  _Float16* Hlv   = (_Float16*)(ws + OFF_HLV);
  float*    Ey    = (float*)(ws + OFF_EY);
  float*    Ey2   = (float*)(ws + OFF_EY2);
  float*    out   = (float*)d_out;

  // prep (also zeroes d_out before the atomics; same stream => ordered)
  k_prep_weights<<<256, 256, 0, stream>>>(muW1, lvW1, muW2, lvW2,
                                          wW1mu, wW1lv, wW2mu, wW2lv, out);
  k_transpose_x<<<(NB * NL * NXD) / 256, 256, 0, stream>>>(x, Xt);
  k_ystats<<<2, 256, 0, stream>>>(y, Ey, Ey2);
  // layer 1 (mu + lv): 4096 waves, one 16x16 tile each
  k_gemm1<<<1024, 128, 0, stream>>>(wW1mu, wW1lv, mub1, lvb1, Xt, Hmu, Hlv);
  // layer 2 + fused loss epilogue: 1024 waves
  k_gemm2<<<256, 128, 0, stream>>>(wW2mu, wW2lv, mub2, lvb2, Hmu, Hlv,
                                   y, Ey, Ey2, out);
}
